// EquivariantLayer_44856638440003
// MI455X (gfx1250) — matrix operand order repaired and checked by reading, not compiled, not run on hardware
//
#include <hip/hip_runtime.h>

// ---------------------------------------------------------------------------
// EquivariantLayer: out = x @ gather(w_table, pattern) + gather(b_table, bias_pattern)
// M=16384, K=2048, N=2048.  bf16 WMMA GEMM, f32 accumulate, double-buffered LDS,
// async global->LDS (ASYNCcnt) for the pre-gathered transposed weight tiles.
// ---------------------------------------------------------------------------

typedef __attribute__((ext_vector_type(16))) __bf16 v16bf;
typedef __attribute__((ext_vector_type(8)))  __bf16 v8bf;
typedef __attribute__((ext_vector_type(4)))  __bf16 v4bf;
typedef __attribute__((ext_vector_type(8)))  float  v8f;
typedef int v4i __attribute__((vector_size(16)));   // matches builtin param type

static constexpr int MDIM = 16384;
static constexpr int KDIM = 2048;
static constexpr int NDIM = 2048;

static constexpr int BM = 128;
static constexpr int BN = 256;
static constexpr int BK = 32;
static constexpr int LDT = BK + 8;      // LDS row stride (80 B = 20 banks -> conflict free)
static constexpr int NT  = KDIM / BK;   // 64 k-steps

#if defined(__has_builtin)
#if __has_builtin(__builtin_amdgcn_global_load_async_to_lds_b128)
#define ASYNC_LDS 1
#endif
#if __has_builtin(__builtin_amdgcn_s_wait_asynccnt)
#define HAVE_WAIT_ASYNC 1
#endif
#endif

union V16 { v16bf v; v8bf h[2]; };

// --- Kernel 1: codebook-gather weight into bf16, stored transposed [N][K] ---
__global__ void gather_wT_kernel(const int* __restrict__ pattern,       // [K][N]
                                 const float* __restrict__ params,      // [64]
                                 __bf16* __restrict__ wT)               // [N][K]
{
    size_t idx = (size_t)blockIdx.x * blockDim.x + threadIdx.x;         // k*N + n
    if (idx >= (size_t)KDIM * NDIM) return;
    int k = (int)(idx / NDIM);
    int n = (int)(idx % NDIM);
    int p = pattern[idx];
    float v = (p == 0) ? 0.0f : params[p - 1];
    wT[(size_t)n * KDIM + k] = (__bf16)v;                               // scatter absorbed by L2 (wT = 8 MB)
}

// --- Kernel 2: codebook-gather bias to f32 ---
__global__ void gather_bias_kernel(const int* __restrict__ bias_pattern, // [N]
                                   const float* __restrict__ bparams,    // [16]
                                   float* __restrict__ bias)             // [N]
{
    int n = blockIdx.x * blockDim.x + threadIdx.x;
    if (n >= NDIM) return;
    int p = bias_pattern[n];
    bias[n] = (p == 0) ? 0.0f : bparams[p - 1];
}

// --- Kernel 3: double-buffered bf16 WMMA GEMM ---
__global__ __launch_bounds__(256)
void gemm_wmma_bf16_kernel(const float* __restrict__ x,     // [M][K] f32
                           const __bf16* __restrict__ wT,   // [N][K] bf16
                           const float* __restrict__ bias,  // [N]
                           float* __restrict__ out)         // [M][N] f32
{
    __shared__ __align__(16) __bf16 Asm[2][BM * LDT];  // x tile,  [row][k]
    __shared__ __align__(16) __bf16 Bsm[2][BN * LDT];  // wT tile, [col][k]

    const int tid  = threadIdx.x;
    const int wave = tid >> 5;
    const int lane = tid & 31;
    const int h    = lane >> 4;      // lane half (ISA 7.12.2 layouts)
    const int lr   = lane & 15;

    const int mb = blockIdx.y * BM;
    const int nb = blockIdx.x * BN;

    const int wrow = (wave & 1) * 64;   // 2 waves along M (64 rows each)
    const int wcol = (wave >> 1) * 64;  // 4 waves along N (64 cols each)

    const v8f vzero = {0.f, 0.f, 0.f, 0.f, 0.f, 0.f, 0.f, 0.f};
    v8f acc[4][4];
#pragma unroll
    for (int mi = 0; mi < 4; ++mi)
#pragma unroll
        for (int nj = 0; nj < 4; ++nj)
            acc[mi][nj] = vzero;

    // ---- staging helpers -------------------------------------------------
    auto loadA_global = [&](float4 (&r)[4], int kb) {
#pragma unroll
        for (int i = 0; i < 4; ++i) {                 // 128 rows x 8 float4 / 256 thr
            int lin = tid + i * 256;
            int row = lin >> 3, cv = lin & 7;
            r[i] = *(const float4*)(x + (size_t)(mb + row) * KDIM + kb + cv * 4);
        }
    };
    auto storeA_lds = [&](int buf, const float4 (&r)[4]) {
#pragma unroll
        for (int i = 0; i < 4; ++i) {
            int lin = tid + i * 256;
            int row = lin >> 3, cv = lin & 7;
            v4bf pk;
            pk.x = (__bf16)r[i].x; pk.y = (__bf16)r[i].y;
            pk.z = (__bf16)r[i].z; pk.w = (__bf16)r[i].w;
            *(v4bf*)&Asm[buf][row * LDT + cv * 4] = pk;
        }
    };
    auto stageB = [&](int buf, int kb) {
#pragma unroll
        for (int i = 0; i < 4; ++i) {                 // 256 rows x 4 16B-chunks / 256 thr
            int lin = tid + i * 256;
            int row = lin >> 2, ch = lin & 3;
            const __bf16* src = wT + (size_t)(nb + row) * KDIM + kb + ch * 8;
            __bf16* dst = &Bsm[buf][row * LDT + ch * 8];
#if ASYNC_LDS
            __builtin_amdgcn_global_load_async_to_lds_b128(
                (__attribute__((address_space(1))) v4i*)src,
                (__attribute__((address_space(3))) v4i*)dst,
                /*imm offset*/0, /*cpol*/0);
#else
            *(v8bf*)dst = *(const v8bf*)src;
#endif
        }
    };
    auto wait_async = [&]() {
#if ASYNC_LDS
#if HAVE_WAIT_ASYNC
        __builtin_amdgcn_s_wait_asynccnt(0);
#else
        asm volatile("s_wait_asynccnt 0x0" ::: "memory");
#endif
#endif
    };
    auto compute = [&](int buf) {
        // A fragments: lane(h, r) holds K = {8h..8h+7} then {16+8h..16+8h+7}
        V16 afrag[4];
#pragma unroll
        for (int mi = 0; mi < 4; ++mi) {
            const __bf16* base = &Asm[buf][(wrow + 16 * mi + lr) * LDT];
            afrag[mi].h[0] = *(const v8bf*)(base + 8 * h);
            afrag[mi].h[1] = *(const v8bf*)(base + 16 + 8 * h);
        }
#pragma unroll
        for (int nj = 0; nj < 4; ++nj) {
            // B fragment: lane(h, c) holds K = {8h..8h+7} then {16+8h..16+8h+7} of column c
            V16 bfrag;
            const __bf16* base = &Bsm[buf][(wcol + 16 * nj + lr) * LDT];
            bfrag.h[0] = *(const v8bf*)(base + 8 * h);
            bfrag.h[1] = *(const v8bf*)(base + 16 + 8 * h);
#pragma unroll
            for (int mi = 0; mi < 4; ++mi)
                acc[mi][nj] = __builtin_amdgcn_wmma_f32_16x16x32_bf16(
                    false, afrag[mi].v, false, bfrag.v,
                    (short)0, acc[mi][nj], false, false);
        }
    };

    // ---- pipelined main loop --------------------------------------------
    {
        float4 r[4];
        loadA_global(r, 0);
        stageB(0, 0);
        storeA_lds(0, r);
        wait_async();
        __syncthreads();
    }

    for (int kt = 0; kt < NT; ++kt) {
        const int cur = kt & 1, nxt = cur ^ 1;
        const bool more = (kt + 1 < NT);
        float4 r[4];
        if (more) {
            loadA_global(r, (kt + 1) * BK);     // global latency hidden under WMMAs
            stageB(nxt, (kt + 1) * BK);         // async DMA straight into LDS
            if (kt + 2 < NT && tid < BM)        // L2 prefetch of x slab t+2
                __builtin_prefetch(x + (size_t)(mb + tid) * KDIM + (kt + 2) * BK, 0, 0);
        }
        compute(cur);
        if (more) storeA_lds(nxt, r);
        wait_async();
        __syncthreads();
    }

    // ---- epilogue: D VGPR j of lane(h,c) = row 8h+j, col c ---------------
#pragma unroll
    for (int nj = 0; nj < 4; ++nj) {
        const int col = nb + wcol + 16 * nj + lr;
        const float bv = bias[col];
#pragma unroll
        for (int mi = 0; mi < 4; ++mi) {
            const int rbase = mb + wrow + 16 * mi + 8 * h;
#pragma unroll
            for (int j = 0; j < 8; ++j)
                out[(size_t)(rbase + j) * NDIM + col] = acc[mi][nj][j] + bv;
        }
    }
}

// ---------------------------------------------------------------------------
extern "C" void kernel_launch(void* const* d_in, const int* in_sizes, int n_in,
                              void* d_out, int out_size, void* d_ws, size_t ws_size,
                              hipStream_t stream) {
    const float* x             = (const float*)d_in[0];   // [16384, 2048]
    const float* matrix_params = (const float*)d_in[1];   // [64]
    const float* bias_params   = (const float*)d_in[2];   // [16]
    const int*   weight_pat    = (const int*)d_in[3];     // [2048, 2048]
    const int*   bias_pat      = (const int*)d_in[4];     // [2048]
    float*       out           = (float*)d_out;           // [16384, 2048]

    // workspace layout: wT bf16 [N][K] (8 MB), then bias f32 [N] (8 KB)
    __bf16* wT   = (__bf16*)d_ws;
    float*  bias = (float*)((char*)d_ws + (size_t)NDIM * KDIM * sizeof(__bf16));

    {
        size_t total = (size_t)KDIM * NDIM;
        int threads = 256;
        int blocks = (int)((total + threads - 1) / threads);
        gather_wT_kernel<<<blocks, threads, 0, stream>>>(weight_pat, matrix_params, wT);
    }
    {
        int threads = 256;
        int blocks = (NDIM + threads - 1) / threads;
        gather_bias_kernel<<<blocks, threads, 0, stream>>>(bias_pat, bias_params, bias);
    }
    {
        dim3 grid(NDIM / BN, MDIM / BM);  // (8, 128)
        gemm_wmma_bf16_kernel<<<grid, 256, 0, stream>>>(x, wT, bias, out);
    }
}
